// Block_2680059592818
// MI455X (gfx1250) — matrix operand order repaired and checked by reading, not compiled, not run on hardware
//
#include <hip/hip_runtime.h>
#include <math.h>

typedef _Float16 half_t;
typedef __attribute__((ext_vector_type(16))) _Float16 v16h;
typedef __attribute__((ext_vector_type(8)))  float    v8f;

#define BB   4
#define HH   128
#define WW   128
#define CC   256
#define LL   (HH*WW)
#define HB   4      // heads per branch
#define HD   32     // head dim
#define CBR  128    // channels per branch
#define NREL 945

// ---------------- fragment loaders (wave32 WMMA 16x16x32 f16) ----------------
// A operand: 16x32, storage row-major A[row][k]:
//   lane<16: row=lane, K=0..15; lane>=16: K=16..31
// -> 16 contiguous halfs per lane (lowers to 2x ds_load_b128)
__device__ inline v16h load_a_frag(const half_t* __restrict__ base, int ld, int lane) {
    int row = lane & 15, hf = lane >> 4;
    const half_t* p = base + row * ld;
    v16h a;
#pragma unroll
    for (int r = 0; r < 8; ++r) { int k = hf * 16 + 2 * r;       a[2*r] = p[k]; a[2*r+1] = p[k+1]; }
    return a;
}

// B operand from transposed storage T[n][k] (stride ld):
//   lane = column n; lanes 0-15 hold K=0..15, lanes 16-31 hold K=16..31
// -> 16 contiguous halfs per lane (lowers to 2x ds_load_b128)
__device__ inline v16h load_bT_frag(const half_t* __restrict__ base, int ld, int lane) {
    int col = lane & 15, koff = (lane >> 4) * 16;
    const half_t* p = base + (size_t)col * ld + koff;
    v16h b;
#pragma unroll
    for (int r = 0; r < 8; ++r) { b[2*r] = p[2*r]; b[2*r+1] = p[2*r+1]; }
    return b;
}

#define WMMA_F16(A, Bv, Cv) \
    __builtin_amdgcn_wmma_f32_16x16x32_f16(false, (A), false, (Bv), (short)0, (Cv), false, false)

// async copy 16 bytes global -> LDS (gfx1250 async path, tracked by ASYNCcnt)
#define ASYNC_CP16(ldsoff, gptr, imm_off)                                        \
    asm volatile("global_load_async_to_lds_b128 %0, %1, off offset:%c2"          \
                 :: "v"(ldsoff), "v"(gptr), "i"(imm_off) : "memory")
#define WAIT_ASYNC() asm volatile("s_wait_asynccnt 0x0" ::: "memory")
#define WAIT_DS()    asm volatile("s_wait_dscnt 0x0" ::: "memory")

// ---------------- elementwise f32 -> f16 ----------------
__global__ void cvt_f32_f16(const float* __restrict__ src, half_t* __restrict__ dst, int n) {
    int i = blockIdx.x * blockDim.x + threadIdx.x;
    int stride = gridDim.x * blockDim.x;
    for (; i < n; i += stride) dst[i] = (half_t)src[i];
}

// f32 [K][N] -> f16 transposed [N][K]
__global__ void cvt_transpose(const float* __restrict__ src, half_t* __restrict__ dst,
                              int K, int N) {
    int idx = blockIdx.x * blockDim.x + threadIdx.x;
    if (idx >= K * N) return;
    int k = idx / N, n = idx % N;
    dst[(size_t)n * K + k] = (half_t)src[idx];
}

// ---------------- tiled WMMA GEMM: C = A(MxK) * Bt(NxK)^T ----------------
// block tile 128x128, 8 waves in 2x4, each wave 64x32 (8 wmma accs)
// double-buffered async global->LDS staging
template <bool OUT_F32>
__global__ __launch_bounds__(256) void gemm_f16(
    const half_t* __restrict__ A, const half_t* __restrict__ Bt,
    float* __restrict__ Cf, half_t* __restrict__ Ch,
    const float* __restrict__ bias, int M, int N, int K) {
    __shared__ __align__(16) half_t As[2][128 * 32];
    __shared__ __align__(16) half_t Bs[2][128 * 32];
    const int t = threadIdx.x, lane = t & 31, wave = t >> 5;
    const int mw = wave >> 2, nw = wave & 3;       // 2 x 4 wave grid
    const int gm = blockIdx.x * 128, gn = blockIdx.y * 128;
    const int row = t >> 1, colh = (t & 1) * 16;   // staging assignment

    auto stage = [&](int buf, int kb) {
        const half_t* ga = A  + (size_t)(gm + row) * K + kb + colh;
        const half_t* gb = Bt + (size_t)(gn + row) * K + kb + colh;
        uint32_t sa = (uint32_t)(uintptr_t)(&As[buf][row * 32 + colh]);
        uint32_t sb = (uint32_t)(uintptr_t)(&Bs[buf][row * 32 + colh]);
        ASYNC_CP16(sa, ga, 0); ASYNC_CP16(sa, ga, 16);
        ASYNC_CP16(sb, gb, 0); ASYNC_CP16(sb, gb, 16);
    };

    v8f acc[4][2] = {};
    stage(0, 0);
    WAIT_ASYNC();
    __syncthreads();
    for (int kb = 0; kb < K; kb += 32) {
        const int cur = (kb >> 5) & 1;
        if (kb + 32 < K) stage(cur ^ 1, kb + 32);   // prefetch overlaps compute
        v16h af[4], bf[2];
#pragma unroll
        for (int ti = 0; ti < 4; ++ti)
            af[ti] = load_a_frag(&As[cur][(mw * 64 + ti * 16) * 32], 32, lane);
#pragma unroll
        for (int tj = 0; tj < 2; ++tj)
            bf[tj] = load_bT_frag(&Bs[cur][(nw * 32 + tj * 16) * 32], 32, lane);
#pragma unroll
        for (int ti = 0; ti < 4; ++ti)
#pragma unroll
            for (int tj = 0; tj < 2; ++tj)
                acc[ti][tj] = WMMA_F16(af[ti], bf[tj], acc[ti][tj]);
        WAIT_ASYNC();
        __syncthreads();
    }
    const int hf = lane >> 4, cn = lane & 15;
#pragma unroll
    for (int ti = 0; ti < 4; ++ti)
#pragma unroll
        for (int tj = 0; tj < 2; ++tj) {
            int col = gn + nw * 32 + tj * 16 + cn;
            size_t base = (size_t)(gm + mw * 64 + ti * 16 + 8 * hf) * N + col;
            if (OUT_F32) {
                float* p = Cf + base;
                float bv = bias[col];
#pragma unroll
                for (int r = 0; r < 8; ++r) p[r * N] = acc[ti][tj][r] + bv;
            } else {
                half_t* p = Ch + base;
#pragma unroll
                for (int r = 0; r < 8; ++r) p[r * N] = (half_t)acc[ti][tj][r];
            }
        }
}

// ---------------- dynamic position bias MLP ----------------
__device__ inline void ln_relu8(const float* in, float* out, const float* g, const float* b) {
    float m = 0.f;
#pragma unroll
    for (int j = 0; j < 8; ++j) m += in[j];
    m *= 0.125f;
    float v = 0.f;
#pragma unroll
    for (int j = 0; j < 8; ++j) { float d = in[j] - m; v += d * d; }
    v *= 0.125f;
    float inv = rsqrtf(v + 1e-5f);
#pragma unroll
    for (int j = 0; j < 8; ++j) { float t = (in[j] - m) * inv * g[j] + b[j]; out[j] = t > 0.f ? t : 0.f; }
}

__global__ void dynpos_kernel(
    const float* __restrict__ pp_w, const float* __restrict__ pp_b,
    const float* __restrict__ g1, const float* __restrict__ b1,
    const float* __restrict__ w1, const float* __restrict__ c1,
    const float* __restrict__ g2, const float* __restrict__ b2,
    const float* __restrict__ w2, const float* __restrict__ c2,
    const float* __restrict__ g3, const float* __restrict__ b3,
    const float* __restrict__ w3, const float* __restrict__ c3,
    float* __restrict__ pos_table) {
    int tid = blockIdx.x * blockDim.x + threadIdx.x;
    if (tid >= 2 * NREL) return;
    int br = tid / NREL, i = tid % NREL;
    int Hsp = br == 0 ? 8 : 32, Wsp = br == 0 ? 32 : 8;
    float dy = (float)(i / (2 * Wsp - 1) + 1 - Hsp);
    float dx = (float)(i % (2 * Wsp - 1) + 1 - Wsp);
    float p[8], q[8];
#pragma unroll
    for (int j = 0; j < 8; ++j)
        p[j] = dy * pp_w[br * 16 + j] + dx * pp_w[br * 16 + 8 + j] + pp_b[br * 8 + j];
    ln_relu8(p, q, g1 + br * 8, b1 + br * 8);
#pragma unroll
    for (int j = 0; j < 8; ++j) {
        float s = c1[br * 8 + j];
        for (int k = 0; k < 8; ++k) s += q[k] * w1[br * 64 + k * 8 + j];
        p[j] = s;
    }
    ln_relu8(p, q, g2 + br * 8, b2 + br * 8);
#pragma unroll
    for (int j = 0; j < 8; ++j) {
        float s = c2[br * 8 + j];
        for (int k = 0; k < 8; ++k) s += q[k] * w2[br * 64 + k * 8 + j];
        p[j] = s;
    }
    ln_relu8(p, q, g3 + br * 8, b3 + br * 8);
#pragma unroll
    for (int h = 0; h < HB; ++h) {
        float s = c3[br * HB + h];
        for (int k = 0; k < 8; ++k) s += q[k] * w3[br * 8 * HB + k * HB + h];
        pos_table[(br * NREL + i) * HB + h] = s;
    }
}

// ---------------- flash-style windowed attention (branch-specialized) ----------------
// one workgroup per (window, head); 8 waves; N=256 tokens, hd=32
template <int BR>
__global__ __launch_bounds__(256) void attn_kernel(
    const half_t* __restrict__ qkv, const float* __restrict__ pos_table,
    float* __restrict__ att) {
    constexpr int Hsp = (BR == 0) ? 8 : 32;
    constexpr int Wsp = (BR == 0) ? 32 : 8;
    constexpr int nWx = WW / Wsp;

    __shared__ __align__(16) half_t Qs[256 * 32];
    __shared__ __align__(16) half_t Ks[256 * 32];
    __shared__ __align__(16) half_t Vt[32 * 256];   // V transposed: [hd][key]
    __shared__ __align__(16) half_t Ps[8][16 * 32];
    __shared__ float PosS[NREL];

    const int wid = blockIdx.x;              // 0..1023
    const int win = wid >> 2, head = wid & 3;
    const int b = win >> 6, w = win & 63;
    const int wy = w / nWx, wx = w % nWx;
    const int t = threadIdx.x, lane = t & 31, wave = t >> 5;
    const int hf = lane >> 4, cn = lane & 15;

    // stage this head's bias table into LDS
    for (int i = t; i < NREL; i += 256)
        PosS[i] = pos_table[(BR * NREL + i) * HB + head];

    {   // stage Q,K (async global->LDS) and V (transposed via VGPRs)
        int n = t;
        int iy = wy * Hsp + n / Wsp, ix = wx * Wsp + n % Wsp;
        size_t rb = ((size_t)b * LL + (size_t)iy * WW + ix) * 768;
        int coff = BR * CBR + head * HD;
        const half_t* gq = qkv + rb + 0   + coff;
        const half_t* gk = qkv + rb + 256 + coff;
        const half_t* gv = qkv + rb + 512 + coff;
        uint32_t sq = (uint32_t)(uintptr_t)(Qs + n * 32);
        uint32_t sk = (uint32_t)(uintptr_t)(Ks + n * 32);
#pragma unroll
        for (int j = 0; j < 4; ++j) {
            ASYNC_CP16(sq, gq, j * 16);
            ASYNC_CP16(sk, gk, j * 16);
        }
        uint4 vtmp[4];
        const uint4* gv4 = (const uint4*)gv;
#pragma unroll
        for (int j = 0; j < 4; ++j) vtmp[j] = gv4[j];
        const half_t* vr = (const half_t*)vtmp;
#pragma unroll
        for (int c = 0; c < 32; ++c) Vt[c * 256 + n] = vr[c];
        WAIT_ASYNC();
    }
    __syncthreads();

    const float scale = 0.17677669529663689f;  // 32^-0.5

    for (int mt2 = 0; mt2 < 2; ++mt2) {
        const int mt = wave * 2 + mt2;
        v16h qa = load_a_frag(Qs + (size_t)mt * 16 * 32, 32, lane);
        const int qib = mt * 16 + 8 * hf;          // rows handled: qib + r, r=0..7
        const int y1 = qib / Wsp, x1b = qib % Wsp; // no wrap: x1b+7 < Wsp for both branches
        float mrow[8], lrow[8];
#pragma unroll
        for (int r = 0; r < 8; ++r) { mrow[r] = -1e30f; lrow[r] = 0.f; }
        v8f acc0 = {}, acc1 = {};

        for (int jb = 0; jb < 8; ++jb) {
            v8f z = {};
            v16h kb0 = load_bT_frag(Ks + (size_t)(jb * 32 + 0)  * 32, 32, lane);
            v16h kb1 = load_bT_frag(Ks + (size_t)(jb * 32 + 16) * 32, 32, lane);
            v8f s0 = WMMA_F16(qa, kb0, z);
            v8f s1 = WMMA_F16(qa, kb1, z);

            const int kj0 = jb * 32 + cn, kj1 = kj0 + 16;
            const int i0b = (y1 - kj0 / Wsp + Hsp - 1) * (2 * Wsp - 1) + (x1b - kj0 % Wsp + Wsp - 1);
            const int i1b = (y1 - kj1 / Wsp + Hsp - 1) * (2 * Wsp - 1) + (x1b - kj1 % Wsp + Wsp - 1);
            float p0[8], p1[8];
#pragma unroll
            for (int r = 0; r < 8; ++r) {
                p0[r] = s0[r] * scale + PosS[i0b + r];
                p1[r] = s1[r] * scale + PosS[i1b + r];
            }
#pragma unroll
            for (int r = 0; r < 8; ++r) {
                float tm = fmaxf(p0[r], p1[r]);
                tm = fmaxf(tm, __shfl_xor(tm, 1, 32));
                tm = fmaxf(tm, __shfl_xor(tm, 2, 32));
                tm = fmaxf(tm, __shfl_xor(tm, 4, 32));
                tm = fmaxf(tm, __shfl_xor(tm, 8, 32));
                float nm = fmaxf(mrow[r], tm);
                float alpha = __expf(mrow[r] - nm);
                mrow[r] = nm;
                float e0 = __expf(p0[r] - nm), e1 = __expf(p1[r] - nm);
                float rs = e0 + e1;
                rs += __shfl_xor(rs, 1, 32);
                rs += __shfl_xor(rs, 2, 32);
                rs += __shfl_xor(rs, 4, 32);
                rs += __shfl_xor(rs, 8, 32);
                lrow[r] = lrow[r] * alpha + rs;
                acc0[r] *= alpha; acc1[r] *= alpha;
                p0[r] = e0; p1[r] = e1;
            }
            // re-layout P (D layout) -> A operand via per-wave LDS scratch
#pragma unroll
            for (int r = 0; r < 8; ++r) {
                int prow = r + 8 * hf;
                Ps[wave][prow * 32 + cn]      = (half_t)p0[r];
                Ps[wave][prow * 32 + 16 + cn] = (half_t)p1[r];
            }
            WAIT_DS();
            v16h pa  = load_a_frag(&Ps[wave][0], 32, lane);
            v16h vb0 = load_bT_frag(Vt + 0 * 256  + jb * 32, 256, lane);
            v16h vb1 = load_bT_frag(Vt + 16 * 256 + jb * 32, 256, lane);
            acc0 = WMMA_F16(pa, vb0, acc0);
            acc1 = WMMA_F16(pa, vb1, acc1);
        }
        // output: window rows qib..qib+7 map to consecutive image pixels (+1 in x)
        {
            int iy = wy * Hsp + y1, ix = wx * Wsp + x1b;
            float* pb = att + ((size_t)b * LL + (size_t)iy * WW + ix) * CC + BR * CBR + head * HD + cn;
#pragma unroll
            for (int r = 0; r < 8; ++r) {
                float invl = 1.f / lrow[r];
                pb[r * CC]      = acc0[r] * invl;
                pb[r * CC + 16] = acc1[r] * invl;
            }
        }
    }
}

// ---------------- LePE depthwise 3x3 + combine -> f16 ----------------
__global__ void lepe_kernel(const half_t* __restrict__ qkv, const float* __restrict__ att,
                            const float* __restrict__ w_dw, const float* __restrict__ b_dw,
                            half_t* __restrict__ y) {
    size_t idx = (size_t)blockIdx.x * blockDim.x + threadIdx.x;
    size_t total = (size_t)BB * LL * CC;
    if (idx >= total) return;
    int c = (int)(idx & 255);
    size_t pl = idx >> 8;
    int b = (int)(pl / LL), l = (int)(pl % LL);
    int iy = l / WW, ix = l % WW;
    float sum = b_dw[c];
#pragma unroll
    for (int dy = -1; dy <= 1; ++dy)
#pragma unroll
        for (int dx = -1; dx <= 1; ++dx) {
            int yy = iy + dy, xx = ix + dx;
            if (yy < 0 || yy >= HH || xx < 0 || xx >= WW) continue;
            float vv = (float)qkv[((size_t)b * LL + (size_t)yy * WW + xx) * 768 + 512 + c];
            sum += vv * w_dw[c * 9 + (dy + 1) * 3 + (dx + 1)];
        }
    y[idx] = (half_t)(att[idx] + sum);
}

// ---------------- launcher ----------------
extern "C" void kernel_launch(void* const* d_in, const int* in_sizes, int n_in,
                              void* d_out, int out_size, void* d_ws, size_t ws_size,
                              hipStream_t stream) {
    (void)in_sizes; (void)n_in; (void)out_size; (void)ws_size;
    const float* x      = (const float*)d_in[0];
    const float* w_qkv  = (const float*)d_in[1];
    const float* w_proj = (const float*)d_in[2];
    const float* b_proj = (const float*)d_in[3];
    const float* w_dw   = (const float*)d_in[4];
    const float* b_dw   = (const float*)d_in[5];
    const float* pp_w   = (const float*)d_in[6];
    const float* pp_b   = (const float*)d_in[7];
    const float* ln1_g  = (const float*)d_in[8];
    const float* ln1_b  = (const float*)d_in[9];
    const float* l1_w   = (const float*)d_in[10];
    const float* l1_b   = (const float*)d_in[11];
    const float* ln2_g  = (const float*)d_in[12];
    const float* ln2_b  = (const float*)d_in[13];
    const float* l2_w   = (const float*)d_in[14];
    const float* l2_b   = (const float*)d_in[15];
    const float* ln3_g  = (const float*)d_in[16];
    const float* ln3_b  = (const float*)d_in[17];
    const float* l3_w   = (const float*)d_in[18];
    const float* l3_b   = (const float*)d_in[19];

    const int M = BB * LL;  // 65536
    char* ws = (char*)d_ws;
    size_t off = 0;
    auto carve = [&](size_t bytes) -> void* {
        void* p = ws + off;
        off = (off + bytes + 255) & ~(size_t)255;
        return p;
    };
    half_t* x_h      = (half_t*)carve((size_t)M * CC * sizeof(half_t));
    half_t* wqkv_t   = (half_t*)carve((size_t)CC * 3 * CC * sizeof(half_t));  // [768][256]
    half_t* wproj_t  = (half_t*)carve((size_t)CC * CC * sizeof(half_t));      // [256][256]
    half_t* qkv_h    = (half_t*)carve((size_t)M * 3 * CC * sizeof(half_t));
    float*  att_f    = (float*)carve((size_t)M * CC * sizeof(float));
    half_t* y_h      = (half_t*)carve((size_t)M * CC * sizeof(half_t));
    float*  pos_t    = (float*)carve((size_t)2 * NREL * HB * sizeof(float));

    // 1) convert inputs to f16 (weights transposed to [N][K])
    cvt_f32_f16<<<4096, 256, 0, stream>>>(x, x_h, M * CC);
    cvt_transpose<<<(CC * 3 * CC + 255) / 256, 256, 0, stream>>>(w_qkv, wqkv_t, CC, 3 * CC);
    cvt_transpose<<<(CC * CC + 255) / 256, 256, 0, stream>>>(w_proj, wproj_t, CC, CC);

    // 2) QKV GEMM: (65536 x 256) @ (256 x 768) -> f16
    gemm_f16<false><<<dim3(M / 128, (3 * CC) / 128), 256, 0, stream>>>(
        x_h, wqkv_t, nullptr, qkv_h, nullptr, M, 3 * CC, CC);

    // 3) dynamic position bias tables
    dynpos_kernel<<<(2 * NREL + 255) / 256, 256, 0, stream>>>(
        pp_w, pp_b, ln1_g, ln1_b, l1_w, l1_b, ln2_g, ln2_b, l2_w, l2_b,
        ln3_g, ln3_b, l3_w, l3_b, pos_t);

    // 4) windowed attention: 2 branch-specialized launches, 256 windows * 4 heads each
    attn_kernel<0><<<1024, 256, 0, stream>>>(qkv_h, pos_t, att_f);
    attn_kernel<1><<<1024, 256, 0, stream>>>(qkv_h, pos_t, att_f);

    // 5) LePE + combine -> f16
    lepe_kernel<<<(M * CC) / 256, 256, 0, stream>>>(qkv_h, att_f, w_dw, b_dw, y_h);

    // 6) output projection: (65536 x 256) @ (256 x 256) + bias -> f32 out
    gemm_f16<true><<<dim3(M / 128, CC / 128), 256, 0, stream>>>(
        y_h, wproj_t, (float*)d_out, nullptr, b_proj, M, CC, CC);
}
